// VectorQuantizer_61383672594728
// MI455X (gfx1250) — matrix operand order repaired and checked by reading, compiled, not run on hardware
//
#include <hip/hip_runtime.h>
#include <hip/hip_bf16.h>

typedef __attribute__((ext_vector_type(16))) __bf16 v16bf;
typedef __attribute__((ext_vector_type(8)))  float  v8f;
typedef __attribute__((ext_vector_type(4)))  int    v4i;

#define D_DIM   512
#define TILE_K  16                 // codes per k-tile
#define CHUNKS  (D_DIM / 32)       // 16 WMMA K-chunks of 32
#define LDS_STRIDE 528             // 512 + 16 pad: rows stay 32B aligned, banks spread

#define AS1 __attribute__((address_space(1)))
#define AS3 __attribute__((address_space(3)))

__device__ __forceinline__ __bf16 tobf(float f) { return (__bf16)f; }

// ---- CDNA5 async global->LDS copy (ASYNCcnt) with inline-asm fallback ------
#if defined(__has_builtin) && __has_builtin(__builtin_amdgcn_global_load_async_to_lds_b128)
#define ASYNC_CP16(gsrc, ldst)                                                     \
    __builtin_amdgcn_global_load_async_to_lds_b128(                                \
        (AS1 v4i*)(void*)(const void*)(gsrc),                                      \
        (AS3 v4i*)(void*)(ldst), 0, 0)
#else
#define ASYNC_CP16(gsrc, ldst)                                                     \
    asm volatile("global_load_async_to_lds_b128 %0, %1, off"                       \
                 :: "v"((unsigned)(unsigned long long)(AS3 void*)(ldst)),          \
                    "v"((const void*)(gsrc))                                       \
                 : "memory")
#endif

#if defined(__has_builtin) && __has_builtin(__builtin_amdgcn_s_wait_asynccnt)
#define WAIT_ASYNC(n) __builtin_amdgcn_s_wait_asynccnt(n)
#else
#define WAIT_ASYNC(n) asm volatile("s_wait_asynccnt %0" :: "i"(n) : "memory")
#endif

// ---------------------------------------------------------------------------
// Kernel 1: per-code squared norms  ||e_k||^2  (wave per row)
// ---------------------------------------------------------------------------
__global__ void __launch_bounds__(256) vq_norm_kernel(const float* __restrict__ emb,
                                                      float* __restrict__ normE) {
    const int row  = blockIdx.x * 8 + (threadIdx.x >> 5);
    const int lane = threadIdx.x & 31;
    float s = 0.0f;
#pragma unroll
    for (int i = 0; i < 4; ++i) {
        const int d = (i * 32 + lane) * 4;
        float4 v = *(const float4*)&emb[(size_t)row * D_DIM + d];
        s += v.x * v.x + v.y * v.y + v.z * v.z + v.w * v.w;
    }
#pragma unroll
    for (int m = 16; m >= 1; m >>= 1) s += __shfl_xor(s, m);
    if (lane == 0) normE[row] = s;
}

// ---------------------------------------------------------------------------
// Kernel 1b: pre-convert codebook fp32 -> bf16 (feeds the async-DMA pipeline)
// ---------------------------------------------------------------------------
__global__ void __launch_bounds__(256) vq_cvt_kernel(const float* __restrict__ emb,
                                                     __bf16* __restrict__ embBf) {
    const size_t i = ((size_t)blockIdx.x * 256 + threadIdx.x) * 8;
    float4 a = *(const float4*)&emb[i];
    float4 b = *(const float4*)&emb[i + 4];
    union { __bf16 h[8]; uint4 u; } pk;
    pk.h[0] = tobf(a.x); pk.h[1] = tobf(a.y); pk.h[2] = tobf(a.z); pk.h[3] = tobf(a.w);
    pk.h[4] = tobf(b.x); pk.h[5] = tobf(b.y); pk.h[6] = tobf(b.z); pk.h[7] = tobf(b.w);
    *(uint4*)&embBf[i] = pk.u;
}

// ---------------------------------------------------------------------------
// Shared helpers for the argmin kernels
// ---------------------------------------------------------------------------
__device__ __forceinline__ void preload_A(const float* __restrict__ flat, int rowA,
                                          int kbase, v16bf* areg) {
#pragma unroll
    for (int ch = 0; ch < CHUNKS; ++ch) {
        const float* p = &flat[(size_t)rowA * D_DIM + ch * 32 + kbase];
        float4 f0 = *(const float4*)(p);
        float4 f1 = *(const float4*)(p + 4);
        float4 f2 = *(const float4*)(p + 8);
        float4 f3 = *(const float4*)(p + 12);
        v16bf a;
        a[0]  = tobf(f0.x); a[1]  = tobf(f0.y); a[2]  = tobf(f0.z); a[3]  = tobf(f0.w);
        a[4]  = tobf(f1.x); a[5]  = tobf(f1.y); a[6]  = tobf(f1.z); a[7]  = tobf(f1.w);
        a[8]  = tobf(f2.x); a[9]  = tobf(f2.y); a[10] = tobf(f2.z); a[11] = tobf(f2.w);
        a[12] = tobf(f3.x); a[13] = tobf(f3.y); a[14] = tobf(f3.z); a[15] = tobf(f3.w);
        areg[ch] = a;
    }
}

__device__ __forceinline__ void write_results(int lane, int m0, float* best, int* bidx,
                                              int* __restrict__ outIdx,
                                              float* __restrict__ outIndsF) {
#pragma unroll
    for (int r = 0; r < 8; ++r) {
        float v  = best[r];
        int   bi = bidx[r];
#pragma unroll
        for (int m = 8; m >= 1; m >>= 1) {
            const float v2 = __shfl_xor(v, m);
            const int   i2 = __shfl_xor(bi, m);
            if (v2 < v || (v2 == v && i2 < bi)) { v = v2; bi = i2; }
        }
        if ((lane & 15) == 0) {
            const int rowW = m0 + r + ((lane >> 4) << 3);  // lane0 -> M=r, lane16 -> M=8+r
            outIdx[rowW]   = bi;
            outIndsF[rowW] = (float)bi;
        }
    }
}

// ---------------------------------------------------------------------------
// Kernel 2a: argmin via bf16 WMMA with CDNA5 async global->LDS double buffer.
// 8 waves/WG, 16 rows/wave (A resident in VGPRs), bf16 codebook streamed by
// GLOBAL_LOAD_ASYNC_TO_LDS_B128 one tile ahead of the WMMAs consuming it.
// ---------------------------------------------------------------------------
__global__ void __launch_bounds__(256) vq_argmin_async_kernel(const float* __restrict__ flat,
                                                              const __bf16* __restrict__ embBf,
                                                              const float* __restrict__ normE,
                                                              int*   __restrict__ outIdx,
                                                              float* __restrict__ outIndsF,
                                                              int K) {
    __shared__ __align__(32) __bf16 smem[2][TILE_K * LDS_STRIDE];

    const int tid    = threadIdx.x;
    const int lane   = tid & 31;
    const int wave   = tid >> 5;
    const int m0     = blockIdx.x * 128 + wave * 16;
    const int rowA   = m0 + (lane & 15);
    const int kbase  = (lane & 16) ? 16 : 0;
    const int col_lo = lane & 15;
    const int bK     = 16 * (lane >> 4);

    v16bf areg[CHUNKS];
    preload_A(flat, rowA, kbase, areg);

    float best[8];
    int   bidx[8];
#pragma unroll
    for (int r = 0; r < 8; ++r) { best[r] = 3.4e38f; bidx[r] = 0; }

    // 4 x 16B async copies per thread fill one 16x512 bf16 tile (16 KB)
    auto issue_tile = [&](__bf16* buf, int kt) {
#pragma unroll
        for (int it = 0; it < 4; ++it) {
            const int c   = it * 256 + tid;        // 1024 chunks of 8 bf16
            const int row = c >> 6;
            const int pos = (c & 63) * 8;
            const __bf16* g = embBf + (size_t)(kt * TILE_K + row) * D_DIM + pos;
            ASYNC_CP16(g, &buf[row * LDS_STRIDE + pos]);
        }
    };

    auto tile_body = [&](const __bf16* __restrict__ buf, int kt) {
        v8f c0 = {};
        v8f c1 = {};
#pragma unroll
        for (int ch = 0; ch < CHUNKS; ch += 2) {
            v16bf b0 = *(const v16bf*)&buf[col_lo * LDS_STRIDE + ch * 32 + bK];
            c0 = __builtin_amdgcn_wmma_f32_16x16x32_bf16(false, areg[ch],     false, b0,
                                                         (short)0, c0, false, false);
            v16bf b1 = *(const v16bf*)&buf[col_lo * LDS_STRIDE + (ch + 1) * 32 + bK];
            c1 = __builtin_amdgcn_wmma_f32_16x16x32_bf16(false, areg[ch + 1], false, b1,
                                                         (short)0, c1, false, false);
        }
        const int   col = kt * TILE_K + col_lo;
        const float nrm = normE[col];
#pragma unroll
        for (int r = 0; r < 8; ++r) {
            const float val = nrm - 2.0f * (c0[r] + c1[r]);
            if (val < best[r]) { best[r] = val; bidx[r] = col; }  // strict <: first index wins
        }
    };

    const int numTiles = K / TILE_K;
    issue_tile(smem[0], 0);
    for (int kt = 0; kt < numTiles - 1; ++kt) {
        __bf16* cur = smem[kt & 1];
        __bf16* nxt = smem[(kt & 1) ^ 1];
        issue_tile(nxt, kt + 1);   // overlap next tile's DMA with this tile's WMMAs
        WAIT_ASYNC(4);             // 4 older copies (cur tile) landed; 4 newer in flight
        __syncthreads();           // cur tile visible to all waves
        tile_body(cur, kt);
        __syncthreads();           // all reads of cur done before it is refilled
    }
    WAIT_ASYNC(0);
    __syncthreads();
    tile_body(smem[(numTiles - 1) & 1], numTiles - 1);

    write_results(lane, m0, best, bidx, outIdx, outIndsF);
}

// ---------------------------------------------------------------------------
// Kernel 2b (fallback when ws can't hold the bf16 codebook): synchronous
// fp32->bf16 staging, identical to the round-1 kernel.
// ---------------------------------------------------------------------------
__global__ void __launch_bounds__(256) vq_argmin_kernel(const float* __restrict__ flat,
                                                        const float* __restrict__ emb,
                                                        const float* __restrict__ normE,
                                                        int*   __restrict__ outIdx,
                                                        float* __restrict__ outIndsF,
                                                        int K) {
    __shared__ __align__(32) __bf16 smem[TILE_K * LDS_STRIDE];

    const int tid    = threadIdx.x;
    const int lane   = tid & 31;
    const int wave   = tid >> 5;
    const int m0     = blockIdx.x * 128 + wave * 16;
    const int rowA   = m0 + (lane & 15);
    const int kbase  = (lane & 16) ? 16 : 0;
    const int col_lo = lane & 15;
    const int bK     = 16 * (lane >> 4);

    v16bf areg[CHUNKS];
    preload_A(flat, rowA, kbase, areg);

    float best[8];
    int   bidx[8];
#pragma unroll
    for (int r = 0; r < 8; ++r) { best[r] = 3.4e38f; bidx[r] = 0; }

    const int numTiles = K / TILE_K;
    for (int kt = 0; kt < numTiles; ++kt) {
#pragma unroll
        for (int it = 0; it < 8; ++it) {
            const int idx = it * 256 + tid;
            const int row = idx >> 7;
            const int d   = (idx & 127) * 4;
            float4 v = *(const float4*)&emb[(size_t)(kt * TILE_K + row) * D_DIM + d];
            union { __bf16 h[4]; uint2 u; } pk;
            pk.h[0] = tobf(v.x); pk.h[1] = tobf(v.y);
            pk.h[2] = tobf(v.z); pk.h[3] = tobf(v.w);
            *(uint2*)&smem[row * LDS_STRIDE + d] = pk.u;
        }
        __syncthreads();

        v8f c0 = {};
        v8f c1 = {};
#pragma unroll
        for (int ch = 0; ch < CHUNKS; ch += 2) {
            v16bf b0 = *(const v16bf*)&smem[col_lo * LDS_STRIDE + ch * 32 + bK];
            c0 = __builtin_amdgcn_wmma_f32_16x16x32_bf16(false, areg[ch],     false, b0,
                                                         (short)0, c0, false, false);
            v16bf b1 = *(const v16bf*)&smem[col_lo * LDS_STRIDE + (ch + 1) * 32 + bK];
            c1 = __builtin_amdgcn_wmma_f32_16x16x32_bf16(false, areg[ch + 1], false, b1,
                                                         (short)0, c1, false, false);
        }
        __syncthreads();

        const int   col = kt * TILE_K + col_lo;
        const float nrm = normE[col];
#pragma unroll
        for (int r = 0; r < 8; ++r) {
            const float val = nrm - 2.0f * (c0[r] + c1[r]);
            if (val < best[r]) { best[r] = val; bidx[r] = col; }
        }
    }

    write_results(lane, m0, best, bidx, outIdx, outIndsF);
}

// ---------------------------------------------------------------------------
// Kernel 3: gather quantized = emb[idx], vq_loss = 1.25 * mean((q-x)^2)
// ---------------------------------------------------------------------------
__global__ void __launch_bounds__(256) vq_gather_kernel(const float* __restrict__ flat,
                                                        const float* __restrict__ emb,
                                                        const int*   __restrict__ inds,
                                                        float* __restrict__ outQ,
                                                        float* __restrict__ outLoss) {
    const int row  = blockIdx.x * 8 + (threadIdx.x >> 5);
    const int lane = threadIdx.x & 31;
    const int idx  = inds[row];
    const float* e = &emb[(size_t)idx * D_DIM];
    const float* x = &flat[(size_t)row * D_DIM];
    float*       q = &outQ[(size_t)row * D_DIM];

    float s = 0.0f;
#pragma unroll
    for (int i = 0; i < 4; ++i) {
        const int d = (i * 32 + lane) * 4;
        float4 ev = *(const float4*)&e[d];
        float4 xv = *(const float4*)&x[d];
        *(float4*)&q[d] = ev;
        const float dx = ev.x - xv.x, dy = ev.y - xv.y;
        const float dz = ev.z - xv.z, dw = ev.w - xv.w;
        s += dx * dx + dy * dy + dz * dz + dw * dw;
    }
#pragma unroll
    for (int m = 16; m >= 1; m >>= 1) s += __shfl_xor(s, m);
    if (lane == 0) outLoss[row] = 1.25f * s * (1.0f / (float)D_DIM);
}

// ---------------------------------------------------------------------------
extern "C" void kernel_launch(void* const* d_in, const int* in_sizes, int n_in,
                              void* d_out, int out_size, void* d_ws, size_t ws_size,
                              hipStream_t stream) {
    const float* latents = (const float*)d_in[0];   // [B,T,D] fp32
    const float* emb     = (const float*)d_in[1];   // [K,D]   fp32

    const int N = in_sizes[0] / D_DIM;              // 16384
    const int K = in_sizes[1] / D_DIM;              // 8192

    float* out      = (float*)d_out;
    float* outQ     = out;                          // N*D  quantized_st
    float* outLoss  = out + (size_t)N * D_DIM;      // N    vq_loss
    float* outIndsF = outLoss + N;                  // N    encoding_inds (as float)

    float*  normE = (float*)d_ws;                                        // K floats
    int*    wsIdx = (int*)((char*)d_ws + (size_t)K * sizeof(float));     // N ints
    __bf16* embBf = (__bf16*)((char*)d_ws + (size_t)K * sizeof(float)
                                          + (size_t)N * sizeof(int));    // K*D bf16

    const size_t need = (size_t)K * sizeof(float) + (size_t)N * sizeof(int)
                      + (size_t)K * D_DIM * sizeof(__bf16);

    vq_norm_kernel<<<K / 8, 256, 0, stream>>>(emb, normE);
    if (ws_size >= need) {
        vq_cvt_kernel<<<(K * D_DIM) / (256 * 8), 256, 0, stream>>>(emb, embBf);
        vq_argmin_async_kernel<<<N / 128, 256, 0, stream>>>(latents, embBf, normE,
                                                            wsIdx, outIndsF, K);
    } else {
        vq_argmin_kernel<<<N / 128, 256, 0, stream>>>(latents, emb, normE,
                                                      wsIdx, outIndsF, K);
    }
    vq_gather_kernel<<<N / 8, 256, 0, stream>>>(latents, emb, wsIdx, outQ, outLoss);
}
